// GAT_81449759801849
// MI455X (gfx1250) — matrix operand order repaired and checked by reading, not compiled, not run on hardware
//
#include <hip/hip_runtime.h>

// ---------------------------------------------------------------------------
// Problem constants (from the reference)
// ---------------------------------------------------------------------------
#define GN    50000      // nodes
#define GE    800000     // edges (before self loops)
#define GETOT 850000     // edges + self loops
#define GIN   128        // input dim
#define GHD   256        // layer width (4 heads x 64)
#define GG    512        // graphs
#define NEG_SLOPE 0.2f

typedef __attribute__((ext_vector_type(16))) __bf16 v16bf;
typedef __attribute__((ext_vector_type(8)))  float  v8f;

union Frag16 { v16bf v; uint4 q[2]; };

// float -> bf16 with round-to-nearest-even (no reliance on scalar __bf16 cvt)
__device__ __forceinline__ unsigned short f2bf(float f) {
  unsigned u = __float_as_uint(f);
  u += 0x7fffu + ((u >> 16) & 1u);
  return (unsigned short)(u >> 16);
}

// order-preserving float<->uint for atomicMax over signed floats
__device__ __forceinline__ unsigned fenc(float f) {
  unsigned u = __float_as_uint(f);
  return (u & 0x80000000u) ? ~u : (u | 0x80000000u);
}
__device__ __forceinline__ float fdec(unsigned u) {
  return __uint_as_float((u & 0x80000000u) ? (u ^ 0x80000000u) : ~u);
}

__device__ __forceinline__ float lrelu(float t) { return t > 0.f ? t : NEG_SLOPE * t; }

// ---------------------------------------------------------------------------
// Utility kernels
// ---------------------------------------------------------------------------
__global__ void zero_f32(float* __restrict__ p, long long n) {
  long long i = (long long)blockIdx.x * blockDim.x + threadIdx.x;
  if (i < n) p[i] = 0.f;
}

__global__ void cvt_f32_bf16(const float* __restrict__ in, unsigned short* __restrict__ out,
                             long long n) {
  long long i = (long long)blockIdx.x * blockDim.x + threadIdx.x;
  if (i < n) out[i] = f2bf(in[i]);
}

// Pre-swizzle W [K, 256] f32 into WMMA B-fragment order (bf16):
// frag(ct, kt): lane l holds 16 contiguous bf16 = B[kt*32 + (l>>4)*16 + i][ct*16 + (l&15)]
__global__ void swizzle_w(const float* __restrict__ W, unsigned short* __restrict__ out, int K) {
  int kTiles = K >> 5;
  int total  = 16 * kTiles * 512;      // ct * kt * (32 lanes * 16 elems)
  int idx = blockIdx.x * blockDim.x + threadIdx.x;
  if (idx >= total) return;
  int i  = idx & 15;
  int l  = (idx >> 4) & 31;
  int t  = idx >> 9;
  int kt = t % kTiles;
  int ct = t / kTiles;
  int k  = kt * 32 + ((l >> 4) << 4) + i;
  int col = ct * 16 + (l & 15);
  out[idx] = f2bf(W[k * GHD + col]);
}

// ---------------------------------------------------------------------------
// GEMM: C[N,256] = A[N,K](bf16) @ Bswz(frag-order bf16), f32 accumulate.
// Block = 256 threads = 8 waves. Wave w: row tile (blockIdx.x*2 + (w>>2)),
// column quad (w&3) -> 4 accumulators of 16x16. kTiles = K/32.
// ---------------------------------------------------------------------------
__global__ void gemm_wmma_bf16(const unsigned short* __restrict__ A,
                               const unsigned short* __restrict__ Bswz,
                               float* __restrict__ C, int Nrows, int K) {
  const int kTiles  = K >> 5;
  const int lane    = threadIdx.x & 31;
  const int w       = threadIdx.x >> 5;
  const int rowTile = blockIdx.x * 2 + (w >> 2);
  const int colQuad = w & 3;
  const int kh      = lane >> 4;
  const int rl      = lane & 15;

  int arow = rowTile * 16 + rl;
  if (arow > Nrows - 1) arow = Nrows - 1;           // clamp (keeps EXEC uniform)
  const unsigned short* abase = A + (size_t)arow * K;

  v8f acc0 = {}, acc1 = {}, acc2 = {}, acc3 = {};

  for (int kt = 0; kt < kTiles; ++kt) {
    Frag16 a;
    a.q[0] = *reinterpret_cast<const uint4*>(abase + kt * 32 + kh * 16);
    a.q[1] = *reinterpret_cast<const uint4*>(abase + kt * 32 + kh * 16 + 8);

    const unsigned short* bb =
        Bswz + (((size_t)((colQuad * 4) * kTiles + kt)) * 32 + lane) * 16;
    const size_t fstride = (size_t)kTiles * 32 * 16;   // elems per column tile

    Frag16 b0, b1, b2, b3;
    b0.q[0] = reinterpret_cast<const uint4*>(bb)[0];
    b0.q[1] = reinterpret_cast<const uint4*>(bb)[1];
    b1.q[0] = reinterpret_cast<const uint4*>(bb + fstride)[0];
    b1.q[1] = reinterpret_cast<const uint4*>(bb + fstride)[1];
    b2.q[0] = reinterpret_cast<const uint4*>(bb + 2 * fstride)[0];
    b2.q[1] = reinterpret_cast<const uint4*>(bb + 2 * fstride)[1];
    b3.q[0] = reinterpret_cast<const uint4*>(bb + 3 * fstride)[0];
    b3.q[1] = reinterpret_cast<const uint4*>(bb + 3 * fstride)[1];

    acc0 = __builtin_amdgcn_wmma_f32_16x16x32_bf16(false, a.v, false, b0.v, (short)0, acc0, false, false);
    acc1 = __builtin_amdgcn_wmma_f32_16x16x32_bf16(false, a.v, false, b1.v, (short)0, acc1, false, false);
    acc2 = __builtin_amdgcn_wmma_f32_16x16x32_bf16(false, a.v, false, b2.v, (short)0, acc2, false, false);
    acc3 = __builtin_amdgcn_wmma_f32_16x16x32_bf16(false, a.v, false, b3.v, (short)0, acc3, false, false);
  }

  // C/D layout: VGPR r -> row = rowTile*16 + r + 8*kh, col = ct*16 + (lane&15)
  const int rowB = rowTile * 16 + 8 * kh;
  const int colB = colQuad * 64 + rl;
  v8f accs[4] = {acc0, acc1, acc2, acc3};
#pragma unroll
  for (int j = 0; j < 4; ++j) {
#pragma unroll
    for (int r = 0; r < 8; ++r) {
      int rr = rowB + r;
      if (rr < Nrows) C[(size_t)rr * GHD + colB + j * 16] = accs[j][r];
    }
  }
}

// ---------------------------------------------------------------------------
// Edge pass 1: logits[e][h] = att . lrelu(xl[src] + xr[dst]); segment max into m.
// One wave per edge; lane handles 8 contiguous channels (head = lane>>3).
// ---------------------------------------------------------------------------
__device__ __forceinline__ float lrelu_dot4(float4 a, float4 l, float4 r) {
  float t0 = lrelu(l.x + r.x), t1 = lrelu(l.y + r.y);
  float t2 = lrelu(l.z + r.z), t3 = lrelu(l.w + r.w);
  return a.x * t0 + a.y * t1 + a.z * t2 + a.w * t3;
}

__global__ void edge_logits(const float* __restrict__ xl, const float* __restrict__ xr,
                            const float* __restrict__ att,
                            const int* __restrict__ esrc, const int* __restrict__ edst,
                            int Eorig, int Etot,
                            float* __restrict__ logits, unsigned* __restrict__ m) {
  int e = blockIdx.x * 8 + (threadIdx.x >> 5);
  if (e >= Etot) return;
  int lane = threadIdx.x & 31;
  int src, dst;
  if (e < Eorig) { src = esrc[e]; dst = edst[e]; } else { src = dst = e - Eorig; }

  int c = lane * 8;
  const float4* pl = reinterpret_cast<const float4*>(xl + (size_t)src * GHD + c);
  const float4* pr = reinterpret_cast<const float4*>(xr + (size_t)dst * GHD + c);
  const float4* pa = reinterpret_cast<const float4*>(att + c);
  float s = lrelu_dot4(pa[0], pl[0], pr[0]) + lrelu_dot4(pa[1], pl[1], pr[1]);

  s += __shfl_xor(s, 1);
  s += __shfl_xor(s, 2);
  s += __shfl_xor(s, 4);

  if ((lane & 7) == 0) {
    int h = lane >> 3;
    logits[(size_t)e * 4 + h] = s;
    atomicMax(&m[(size_t)dst * 4 + h], fenc(s));
  }
}

// ---------------------------------------------------------------------------
// Edge pass 2: p = exp(logit - m[dst]); denom += p; aggr[dst] += p * xl[src].
// ---------------------------------------------------------------------------
__global__ void edge_aggr(const float* __restrict__ xl, const float* __restrict__ logits,
                          const unsigned* __restrict__ m,
                          const int* __restrict__ esrc, const int* __restrict__ edst,
                          int Eorig, int Etot,
                          float* __restrict__ denom, float* __restrict__ aggr) {
  int e = blockIdx.x * 8 + (threadIdx.x >> 5);
  if (e >= Etot) return;
  int lane = threadIdx.x & 31;
  int src, dst;
  if (e < Eorig) { src = esrc[e]; dst = edst[e]; } else { src = dst = e - Eorig; }

  int h = lane >> 3;
  float logit = logits[(size_t)e * 4 + h];
  float mm = fdec(m[(size_t)dst * 4 + h]);
  float p = __expf(logit - mm);
  if ((lane & 7) == 0) atomicAdd(&denom[(size_t)dst * 4 + h], p);

  int c = lane * 8;
  const float4* pl = reinterpret_cast<const float4*>(xl + (size_t)src * GHD + c);
  float4 l0 = pl[0], l1 = pl[1];
  float* d = aggr + (size_t)dst * GHD + c;
  atomicAdd(d + 0, p * l0.x); atomicAdd(d + 1, p * l0.y);
  atomicAdd(d + 2, p * l0.z); atomicAdd(d + 3, p * l0.w);
  atomicAdd(d + 4, p * l1.x); atomicAdd(d + 5, p * l1.y);
  atomicAdd(d + 6, p * l1.z); atomicAdd(d + 7, p * l1.w);
}

// h = elu(aggr/denom + bias), in place
__global__ void node_finalize(float* __restrict__ hbuf, const float* __restrict__ denom,
                              const float* __restrict__ bias, long long total) {
  long long idx = (long long)blockIdx.x * blockDim.x + threadIdx.x;
  if (idx >= total) return;
  int n = (int)(idx >> 8);
  int c = (int)(idx & 255);
  int h = c >> 6;
  float v = hbuf[idx] / denom[n * 4 + h] + bias[c];
  hbuf[idx] = v > 0.f ? v : (__expf(v) - 1.f);
}

// ---------------------------------------------------------------------------
// Pooling + head
// ---------------------------------------------------------------------------
__global__ void pool_sum(const float* __restrict__ h, const int* __restrict__ batch,
                         float* __restrict__ sums, long long total) {
  long long idx = (long long)blockIdx.x * blockDim.x + threadIdx.x;
  if (idx >= total) return;
  int n = (int)(idx >> 8);
  int c = (int)(idx & 255);
  atomicAdd(&sums[(size_t)batch[n] * GHD + c], h[idx]);
}

__global__ void pool_count(const int* __restrict__ batch, float* __restrict__ counts, int n) {
  int i = blockIdx.x * blockDim.x + threadIdx.x;
  if (i < n) atomicAdd(&counts[batch[i]], 1.f);
}

__global__ void head_kernel(const float* __restrict__ sums, const float* __restrict__ counts,
                            const float* __restrict__ headW, const float* __restrict__ headb,
                            float* __restrict__ out, int G) {
  int g = blockIdx.x * 8 + (threadIdx.x >> 5);
  if (g >= G) return;
  int lane = threadIdx.x & 31;
  const float4* ps = reinterpret_cast<const float4*>(sums + (size_t)g * GHD + lane * 8);
  const float4* pw = reinterpret_cast<const float4*>(headW + lane * 8);
  float4 s0 = ps[0], s1 = ps[1], w0 = pw[0], w1 = pw[1];
  float s = s0.x * w0.x + s0.y * w0.y + s0.z * w0.z + s0.w * w0.w +
            s1.x * w1.x + s1.y * w1.y + s1.z * w1.z + s1.w * w1.w;
  s += __shfl_xor(s, 1);  s += __shfl_xor(s, 2);  s += __shfl_xor(s, 4);
  s += __shfl_xor(s, 8);  s += __shfl_xor(s, 16);
  if (lane == 0) out[g] = s / fmaxf(counts[g], 1.f) + headb[0];
}

// ---------------------------------------------------------------------------
// Launch
// ---------------------------------------------------------------------------
extern "C" void kernel_launch(void* const* d_in, const int* in_sizes, int n_in,
                              void* d_out, int out_size, void* d_ws, size_t ws_size,
                              hipStream_t stream) {
  (void)in_sizes; (void)n_in; (void)out_size; (void)ws_size;

  const float* x      = (const float*)d_in[0];
  const int*   ei     = (const int*)d_in[1];          // [2, E]
  const int*   batch  = (const int*)d_in[2];
  const float* Wl[3]  = {(const float*)d_in[3], (const float*)d_in[7],  (const float*)d_in[11]};
  const float* Wr[3]  = {(const float*)d_in[4], (const float*)d_in[8],  (const float*)d_in[12]};
  const float* att[3] = {(const float*)d_in[5], (const float*)d_in[9],  (const float*)d_in[13]};
  const float* bia[3] = {(const float*)d_in[6], (const float*)d_in[10], (const float*)d_in[14]};
  const float* headW  = (const float*)d_in[15];
  const float* headb  = (const float*)d_in[16];
  float* out = (float*)d_out;

  char* ws = (char*)d_ws;
  size_t off = 0;
  auto alloc = [&](size_t bytes) -> char* {
    char* p = ws + off;
    off += (bytes + 255) & ~(size_t)255;
    return p;
  };

  float*          hbuf   = (float*)alloc((size_t)GN * GHD * 4);   // h / aggr (aliased)
  float*          xl     = (float*)alloc((size_t)GN * GHD * 4);
  float*          xr     = (float*)alloc((size_t)GN * GHD * 4);
  unsigned short* hbf    = (unsigned short*)alloc((size_t)GN * GHD * 2);
  float*          logits = (float*)alloc((size_t)GETOT * 4 * 4);
  unsigned*       mbuf   = (unsigned*)alloc((size_t)GN * 4 * 4);
  float*          denom  = (float*)alloc((size_t)GN * 4 * 4);
  unsigned short* wlswz  = (unsigned short*)alloc((size_t)GHD * GHD * 2);
  unsigned short* wrswz  = (unsigned short*)alloc((size_t)GHD * GHD * 2);
  float*          sums   = (float*)alloc((size_t)GG * GHD * 4);
  float*          counts = (float*)alloc((size_t)GG * 4);

  const int* esrc = ei;
  const int* edst = ei + GE;

  const int rowTiles = GN / 16;                 // 3125
  const int gemmGrid = (rowTiles + 1) / 2;      // 2 row tiles per block
  const int egrid    = (GETOT + 7) / 8;         // 8 edges (waves) per block
  const long long nodeElems = (long long)GN * GHD;

  const float* hin = x;
  int K = GIN;
  for (int l = 0; l < 3; ++l) {
    long long nk = (long long)GN * K;
    cvt_f32_bf16<<<(unsigned)((nk + 255) / 256), 256, 0, stream>>>(hin, hbf, nk);

    int kTiles = K / 32;
    int swTot  = 16 * kTiles * 512;
    swizzle_w<<<(swTot + 255) / 256, 256, 0, stream>>>(Wl[l], wlswz, K);
    swizzle_w<<<(swTot + 255) / 256, 256, 0, stream>>>(Wr[l], wrswz, K);

    gemm_wmma_bf16<<<gemmGrid, 256, 0, stream>>>(hbf, wlswz, xl, GN, K);
    gemm_wmma_bf16<<<gemmGrid, 256, 0, stream>>>(hbf, wrswz, xr, GN, K);

    zero_f32<<<(unsigned)((nodeElems + 255) / 256), 256, 0, stream>>>(hbuf, nodeElems);
    zero_f32<<<(GN * 4 + 255) / 256, 256, 0, stream>>>(denom, (long long)GN * 4);
    zero_f32<<<(GN * 4 + 255) / 256, 256, 0, stream>>>((float*)mbuf, (long long)GN * 4);

    edge_logits<<<egrid, 256, 0, stream>>>(xl, xr, att[l], esrc, edst, GE, GETOT, logits, mbuf);
    edge_aggr  <<<egrid, 256, 0, stream>>>(xl, logits, mbuf, esrc, edst, GE, GETOT, denom, hbuf);

    node_finalize<<<(unsigned)((nodeElems + 255) / 256), 256, 0, stream>>>(hbuf, denom, bia[l],
                                                                            nodeElems);
    hin = hbuf;
    K = GHD;
  }

  zero_f32<<<(GG * GHD + 255) / 256, 256, 0, stream>>>(sums, (long long)GG * GHD);
  zero_f32<<<(GG + 255) / 256, 256, 0, stream>>>(counts, (long long)GG);
  pool_sum<<<(unsigned)((nodeElems + 255) / 256), 256, 0, stream>>>(hbuf, batch, sums, nodeElems);
  pool_count<<<(GN + 255) / 256, 256, 0, stream>>>(batch, counts, GN);
  head_kernel<<<(GG + 7) / 8, 256, 0, stream>>>(sums, counts, headW, headb, out, GG);
}
